// AdaptiveGraphNetwork_50568944943487
// MI455X (gfx1250) — compile-verified, matrix operand order checked by reading, not run on hardware
//
#include <hip/hip_runtime.h>
#include <hip/hip_bf16.h>
#include <math.h>

// ---------------------------------------------------------------------------
// AdaptiveGraphNetwork forward for gfx1250 (MI455X), wave32 + WMMA f16.
// B operands stored in LDS pre-swizzled into WMMA fragment order so each
// fragment load is 2x ds_load_b128 instead of 16x ds_load_u16.
// ---------------------------------------------------------------------------

typedef __attribute__((ext_vector_type(16))) _Float16 v16h;
typedef __attribute__((ext_vector_type(8)))  _Float16 v8h;
typedef __attribute__((ext_vector_type(8)))  float    v8f;

#define N_NODES 100000
#define N_EDGES 1600000
#define IN_DIM  7
#define HID     128

#define WMMA_F16(a, b, c) \
  __builtin_amdgcn_wmma_f32_16x16x32_f16(false, (a), false, (b), (short)0, (c), false, false)

#define CAT16(lo, hi) \
  __builtin_shufflevector((lo), (hi), 0, 1, 2, 3, 4, 5, 6, 7, 8, 9, 10, 11, 12, 13, 14, 15)

// A fragment: 16x32 (MxK) f16 tile, row-major source, leading dim ld (elements).
// ISA 7.12.2: lane m (0-15) row m holds K = {kb..kb+7, kb+16..kb+23}, kb = 8*(lane>>4).
// Both chunks are 16B-contiguous & aligned -> 2x ds_load_b128.
static __device__ __forceinline__ v16h frag_a(const _Float16* base, int ld, int k0, int lane) {
  const int row = lane & 15;
  const int kb  = k0 + ((lane >> 4) << 3);
  const _Float16* p = base + row * ld + kb;
  v8h lo = *(const v8h*)(p);
  v8h hi = *(const v8h*)(p + 16);
  return CAT16(lo, hi);
}

// B fragment from LDS in pre-swizzled fragment order:
// layout [ks][colblock(8)][lane(32)][16 halves] -> one 32B contiguous run per lane.
static __device__ __forceinline__ v16h frag_b_sw(const _Float16* base, int ks, int cb, int lane) {
  const _Float16* p = base + ((((ks << 3) + cb) << 5) + lane) * 16;
  v8h lo = *(const v8h*)(p);
  v8h hi = *(const v8h*)(p + 8);
  return CAT16(lo, hi);
}

// Stage a row-major f32 weight W[kdim x HID] into LDS f16 in fragment order.
static __device__ __forceinline__ void stage_b_sw(_Float16* dst, const float* __restrict__ src,
                                                  int kdim, int tid) {
  const int total = kdim * HID;
  for (int i = tid; i < total; i += 256) {
    int j    = i & 15;         // half index within lane fragment
    int lane = (i >> 4) & 31;
    int cb   = (i >> 9) & 7;   // column block (16 cols each)
    int ks   = i >> 12;        // K step (32 rows each)
    int k = ks * 32 + ((lane >> 4) << 3) + (j & 7) + ((j >> 3) << 4);
    int n = cb * 16 + (lane & 15);
    dst[i] = (_Float16)src[k * HID + n];
  }
}

static __device__ __forceinline__ float atomicMaxF(float* addr, float val) {
  int* ia  = (int*)addr;
  int  old = __float_as_int(*addr);
  while (__int_as_float(old) < val) {
    int assumed = old;
    old = atomicCAS(ia, assumed, __float_as_int(val));
    if (old == assumed) break;
  }
  return __int_as_float(old);
}

// ---------------------------------------------------------------------------

__global__ void k_fill_neginf(float* p, int n) {
  int i = blockIdx.x * blockDim.x + threadIdx.x;
  if (i < n) p[i] = -INFINITY;
}

// SAGE1 aggregation: scatter x[src] (7 feats) + degree into dst.
__global__ void k_sage1_agg(const float* __restrict__ x, const int* __restrict__ ei,
                            float* __restrict__ agg1, float* __restrict__ deg) {
  int e = blockIdx.x * blockDim.x + threadIdx.x;
  if (e >= N_EDGES) return;
  int s = ei[e], d = ei[N_EDGES + e];
  const float* xs = x + s * IN_DIM;
  float* ad = agg1 + d * IN_DIM;
#pragma unroll
  for (int k = 0; k < IN_DIM; ++k) atomicAdd(&ad[k], xs[k]);
  atomicAdd(&deg[d], 1.0f);
}

// SAGE1 combine: h1 = relu(mean @ w1_l + b1 + x @ w1_r). 7-dim K: plain VALU.
__global__ void k_sage1_combine(const float* __restrict__ x, const float* __restrict__ agg1,
                                const float* __restrict__ deg,
                                const float* __restrict__ w1l, const float* __restrict__ b1,
                                const float* __restrict__ w1r, float* __restrict__ h1) {
  int idx = blockIdx.x * blockDim.x + threadIdx.x;
  if (idx >= N_NODES * HID) return;
  int n = idx >> 7, o = idx & 127;
  float dg = deg[n]; dg = dg < 1.0f ? 1.0f : dg;
  float s = b1[o];
#pragma unroll
  for (int k = 0; k < IN_DIM; ++k) {
    float mk = agg1[n * IN_DIM + k] / dg;
    s += mk * w1l[k * HID + o] + x[n * IN_DIM + k] * w1r[k * HID + o];
  }
  h1[idx] = s > 0.f ? s : 0.f;
}

// SAGE2 aggregation: one wave per edge, each lane scatters 4 floats (L2 atomics).
__global__ void k_sage2_agg(const int* __restrict__ ei, const float* __restrict__ h1,
                            float* __restrict__ agg2) {
  int gt = blockIdx.x * blockDim.x + threadIdx.x;
  int e  = gt >> 5;
  if (e >= N_EDGES) return;
  int lane = gt & 31;
  int s = ei[e], d = ei[N_EDGES + e];
  const float4 v = *(const float4*)(h1 + s * HID + lane * 4);
  float* p = agg2 + d * HID + lane * 4;
  atomicAdd(p + 0, v.x); atomicAdd(p + 1, v.y);
  atomicAdd(p + 2, v.z); atomicAdd(p + 3, v.w);
}

// SAGE2 combine + vertex MLP + GAT node scalars. 16 nodes/block, 8 waves.
__global__ void __launch_bounds__(256, 1) k_node_wmma(
    const float* __restrict__ agg2, const float* __restrict__ deg,
    const float* __restrict__ h1,
    const float* __restrict__ w2l, const float* __restrict__ b2, const float* __restrict__ w2r,
    const float* __restrict__ wv1, const float* __restrict__ bv1,
    const float* __restrict__ wv2, const float* __restrict__ bv2,
    const float* __restrict__ wg, const float* __restrict__ att_src,
    const float* __restrict__ att_dst,
    _Float16* __restrict__ h2f16, float* __restrict__ hN,
    float* __restrict__ a_s, float* __restrict__ a_d,
    float* __restrict__ pos_out) {
  __shared__ __align__(16) _Float16 sW2L[HID * HID];  // fragment order
  __shared__ __align__(16) _Float16 sW2R[HID * HID];  // fragment order
  __shared__ __align__(16) _Float16 sWV1[HID * HID];  // fragment order
  __shared__ __align__(16) _Float16 sAm[16 * HID];
  __shared__ __align__(16) _Float16 sAh[16 * HID];
  __shared__ __align__(16) _Float16 sH2[16 * HID];
  __shared__ __align__(16) _Float16 sT1[16 * HID];
  __shared__ float sRed[256];

  const int tid = threadIdx.x;
  const int wave = tid >> 5, lane = tid & 31;
  const int node0 = blockIdx.x * 16;

  stage_b_sw(sW2L, w2l, HID, tid);
  stage_b_sw(sW2R, w2r, HID, tid);
  stage_b_sw(sWV1, wv1, HID, tid);
  for (int i = tid; i < 16 * HID; i += 256) {
    int r = i >> 7, c = i & 127;
    int n = node0 + r;
    float dg = deg[n]; dg = dg < 1.f ? 1.f : dg;
    sAm[i] = (_Float16)(agg2[n * HID + c] / dg);
    sAh[i] = (_Float16)h1[n * HID + c];
  }
  __syncthreads();

  v8f acc = {};
#pragma unroll
  for (int k = 0; k < 4; ++k) {
    v16h a = frag_a(sAm, HID, k * 32, lane);
    v16h b = frag_b_sw(sW2L, k, wave, lane);
    acc = WMMA_F16(a, b, acc);
  }
#pragma unroll
  for (int k = 0; k < 4; ++k) {
    v16h a = frag_a(sAh, HID, k * 32, lane);
    v16h b = frag_b_sw(sW2R, k, wave, lane);
    acc = WMMA_F16(a, b, acc);
  }
  {
    const int n = (lane & 15) + wave * 16;
    const float bn = b2[n];
#pragma unroll
    for (int r = 0; r < 8; ++r) {
      int m = r + ((lane >> 4) << 3);
      float v = acc[r] + bn;
      _Float16 hv = (_Float16)v;
      sH2[m * HID + n] = hv;
      h2f16[(node0 + m) * HID + n] = hv;
    }
  }
  __syncthreads();

  {  // GAT per-node scalars: hN = h2 . wg  (256-thread partials, 16-thread combine)
    int er = tid >> 4, c0 = tid & 15;
    float p = 0.f;
#pragma unroll
    for (int c = c0; c < HID; c += 16) p += (float)sH2[er * HID + c] * wg[c];
    sRed[tid] = p;
  }
  __syncthreads();
  if (tid < 16) {
    float s = 0.f;
#pragma unroll
    for (int j = 0; j < 16; ++j) s += sRed[tid * 16 + j];
    int n = node0 + tid;
    hN[n]  = s;
    a_s[n] = s * att_src[0];
    a_d[n] = s * att_dst[0];
  }

  v8f acc2 = {};
#pragma unroll
  for (int k = 0; k < 4; ++k) {
    v16h a = frag_a(sH2, HID, k * 32, lane);
    v16h b = frag_b_sw(sWV1, k, wave, lane);
    acc2 = WMMA_F16(a, b, acc2);
  }
  {
    const int n = (lane & 15) + wave * 16;
    const float bn = bv1[n];
#pragma unroll
    for (int r = 0; r < 8; ++r) {
      int m = r + ((lane >> 4) << 3);
      float v = acc2[r] + bn;
      sT1[m * HID + n] = (_Float16)(v > 0.f ? v : 0.f);
    }
  }
  __syncthreads();

  // Vertex layer 2: [16 x 128] @ wv2[128 x 3]. 192 partial threads, 48 finishers.
  if (tid < 192) {
    int r = tid / 12, rem = tid % 12;
    int o = rem >> 2, seg = rem & 3;
    float p = 0.f;
#pragma unroll
    for (int c = seg * 32; c < seg * 32 + 32; ++c)
      p += (float)sT1[r * HID + c] * wv2[c * 3 + o];
    sRed[tid] = p;
  }
  __syncthreads();
  if (tid < 48) {
    int r = tid / 3, o = tid - r * 3;
    float s = bv2[o] + sRed[r * 12 + o * 4] + sRed[r * 12 + o * 4 + 1] +
              sRed[r * 12 + o * 4 + 2] + sRed[r * 12 + o * 4 + 3];
    pos_out[(node0 + r) * 3 + o] = s;
  }
}

// ---- GAT edge softmax passes ----
__global__ void k_gat_max(const int* __restrict__ ei, const float* __restrict__ a_s,
                          const float* __restrict__ a_d, float* __restrict__ mmax) {
  int e = blockIdx.x * blockDim.x + threadIdx.x;
  if (e >= N_EDGES) return;
  int s = ei[e], d = ei[N_EDGES + e];
  float v = a_s[s] + a_d[d];
  v = v > 0.f ? v : 0.2f * v;
  atomicMaxF(&mmax[d], v);
}

__global__ void k_gat_den(const int* __restrict__ ei, const float* __restrict__ a_s,
                          const float* __restrict__ a_d, const float* __restrict__ mmax,
                          float* __restrict__ den) {
  int e = blockIdx.x * blockDim.x + threadIdx.x;
  if (e >= N_EDGES) return;
  int s = ei[e], d = ei[N_EDGES + e];
  float v = a_s[s] + a_d[d];
  v = v > 0.f ? v : 0.2f * v;
  atomicAdd(&den[d], __expf(v - mmax[d]));
}

__global__ void k_gat_acc(const int* __restrict__ ei, const float* __restrict__ a_s,
                          const float* __restrict__ a_d, const float* __restrict__ mmax,
                          const float* __restrict__ den, const float* __restrict__ hN,
                          float* __restrict__ gacc) {
  int e = blockIdx.x * blockDim.x + threadIdx.x;
  if (e >= N_EDGES) return;
  int s = ei[e], d = ei[N_EDGES + e];
  float v = a_s[s] + a_d[d];
  v = v > 0.f ? v : 0.2f * v;
  float alpha = __expf(v - mmax[d]) / (den[d] + 1e-16f);
  atomicAdd(&gacc[d], alpha * hN[s]);
}

__global__ void k_gat_fin(const float* __restrict__ gacc, const float* __restrict__ bg,
                          float* __restrict__ ew_out) {
  int n = blockIdx.x * blockDim.x + threadIdx.x;
  if (n < N_NODES) ew_out[n] = gacc[n] + bg[0];
}

// Edge MLP: sigmoid(relu([h2[src]|h2[dst]] @ we1 + be1) @ we2 + be2).
// 64 edges per block (4 tiles of 16) amortize the 64KB we1 LDS stage.
__global__ void __launch_bounds__(256, 1) k_edge_mlp(
    const int* __restrict__ ei, const _Float16* __restrict__ h2f16,
    const float* __restrict__ we1, const float* __restrict__ be1,
    const float* __restrict__ we2, const float* __restrict__ be2,
    float* __restrict__ conn_out) {
  __shared__ __align__(16) _Float16 sW[256 * HID];  // 64KB we1 f16, fragment order
  __shared__ __align__(16) _Float16 sA[16 * 256];   // 8KB  edge-feature tile
  __shared__ __align__(16) float    sT[16 * HID];   // 8KB  weighted hidden
  __shared__ float sRed[256];

  const int tid = threadIdx.x;
  const int wave = tid >> 5, lane = tid & 31;

  stage_b_sw(sW, we1, 256, tid);

  for (int t = 0; t < 4; ++t) {
    const int e0 = blockIdx.x * 64 + t * 16;
    __syncthreads();
    {  // gather 16 edge rows: [h2f16[src] | h2f16[dst]] as 16B chunks
      int er = tid >> 4;
      int ch = tid & 15;
      int e = e0 + er;
      int s = ei[e], d = ei[N_EDGES + e];
      const uint4* hs = (const uint4*)(h2f16 + s * HID);
      const uint4* hd = (const uint4*)(h2f16 + d * HID);
      uint4* Ar = (uint4*)(sA + er * 256);
      Ar[ch]      = hs[ch];
      Ar[16 + ch] = hd[ch];
    }
    __syncthreads();
    v8f acc = {};
#pragma unroll
    for (int ks = 0; ks < 8; ++ks) {
      v16h a = frag_a(sA, 256, ks * 32, lane);
      v16h b = frag_b_sw(sW, ks, wave, lane);
      acc = WMMA_F16(a, b, acc);
    }
    {
      const int n = (lane & 15) + wave * 16;
      const float bn = be1[n], w2 = we2[n];
#pragma unroll
      for (int r = 0; r < 8; ++r) {
        int m = r + ((lane >> 4) << 3);
        float v = acc[r] + bn;
        v = v > 0.f ? v : 0.f;
        sT[m * HID + n] = v * w2;
      }
    }
    __syncthreads();
    {  // parallel reduction of 128 weighted hiddens per edge
      int er = tid >> 4, c0 = tid & 15;
      float p = 0.f;
#pragma unroll
      for (int c = c0; c < HID; c += 16) p += sT[er * HID + c];
      sRed[tid] = p;
    }
    __syncthreads();
    if (tid < 16) {
      float s = be2[0];
#pragma unroll
      for (int j = 0; j < 16; ++j) s += sRed[tid * 16 + j];
      conn_out[e0 + tid] = 1.f / (1.f + __expf(-s));
    }
  }
}

// ---------------------------------------------------------------------------

extern "C" void kernel_launch(void* const* d_in, const int* in_sizes, int n_in,
                              void* d_out, int out_size, void* d_ws, size_t ws_size,
                              hipStream_t stream) {
  (void)in_sizes; (void)n_in; (void)out_size; (void)ws_size;

  const float* x       = (const float*)d_in[0];
  const int*   ei      = (const int*)d_in[1];
  // d_in[2] edge_attr: unused in forward
  const float* w1l     = (const float*)d_in[3];
  const float* b1      = (const float*)d_in[4];
  const float* w1r     = (const float*)d_in[5];
  const float* w2l     = (const float*)d_in[6];
  const float* b2      = (const float*)d_in[7];
  const float* w2r     = (const float*)d_in[8];
  const float* wg      = (const float*)d_in[9];
  const float* att_src = (const float*)d_in[10];
  const float* att_dst = (const float*)d_in[11];
  const float* bg      = (const float*)d_in[12];
  const float* wv1     = (const float*)d_in[13];
  const float* bv1     = (const float*)d_in[14];
  const float* wv2     = (const float*)d_in[15];
  const float* bv2     = (const float*)d_in[16];
  const float* we1     = (const float*)d_in[17];
  const float* be1     = (const float*)d_in[18];
  const float* we2     = (const float*)d_in[19];
  const float* be2     = (const float*)d_in[20];

  // Workspace layout (floats): [deg N][agg1 7N][agg2 128N][h1 128N]
  //                            [h2f16 128N halves][hN N][a_s N][a_d N][mmax N][den N][gacc N]
  float* ws   = (float*)d_ws;
  float* deg  = ws;
  float* agg1 = deg  + N_NODES;
  float* agg2 = agg1 + (size_t)N_NODES * IN_DIM;
  float* h1   = agg2 + (size_t)N_NODES * HID;
  _Float16* h2f16 = (_Float16*)(h1 + (size_t)N_NODES * HID);
  float* hNb  = (float*)(h2f16 + (size_t)N_NODES * HID);
  float* a_s  = hNb  + N_NODES;
  float* a_d  = a_s  + N_NODES;
  float* mmax = a_d  + N_NODES;
  float* den  = mmax + N_NODES;
  float* gacc = den  + N_NODES;

  float* pos_out  = (float*)d_out;                  // [N,3]
  float* conn_out = pos_out + (size_t)N_NODES * 3;  // [E,1]
  float* ew_out   = conn_out + (size_t)N_EDGES;     // [N,1]

  // Zero accumulators (deg|agg1|agg2 contiguous; den|gacc contiguous).
  hipMemsetAsync(deg, 0, sizeof(float) * ((size_t)N_NODES * (1 + IN_DIM + HID)), stream);
  hipMemsetAsync(den, 0, sizeof(float) * (size_t)N_NODES * 2, stream);
  k_fill_neginf<<<(N_NODES + 255) / 256, 256, 0, stream>>>(mmax, N_NODES);

  // SAGE layer 1
  k_sage1_agg<<<(N_EDGES + 255) / 256, 256, 0, stream>>>(x, ei, agg1, deg);
  k_sage1_combine<<<(N_NODES * HID) / 256, 256, 0, stream>>>(x, agg1, deg, w1l, b1, w1r, h1);

  // SAGE layer 2 aggregation (wave per edge)
  k_sage2_agg<<<(N_EDGES * 32) / 256, 256, 0, stream>>>(ei, h1, agg2);

  // SAGE2 combine + vertex MLP + GAT node scalars (WMMA)
  k_node_wmma<<<N_NODES / 16, 256, 0, stream>>>(
      agg2, deg, h1, w2l, b2, w2r, wv1, bv1, wv2, bv2,
      wg, att_src, att_dst, h2f16, hNb, a_s, a_d, pos_out);

  // GAT edge softmax
  k_gat_max<<<(N_EDGES + 255) / 256, 256, 0, stream>>>(ei, a_s, a_d, mmax);
  k_gat_den<<<(N_EDGES + 255) / 256, 256, 0, stream>>>(ei, a_s, a_d, mmax, den);
  k_gat_acc<<<(N_EDGES + 255) / 256, 256, 0, stream>>>(ei, a_s, a_d, mmax, den, hNb, gacc);
  k_gat_fin<<<(N_NODES + 255) / 256, 256, 0, stream>>>(gacc, bg, ew_out);

  // Edge MLP (WMMA)
  k_edge_mlp<<<N_EDGES / 64, 256, 0, stream>>>(ei, h2f16, we1, be1, we2, be2, conn_out);
}
